// AttnNet_50852412784797
// MI455X (gfx1250) — matrix operand order repaired and checked by reading, not compiled, not run on hardware
//
#include <hip/hip_runtime.h>
#include <hip/hip_bf16.h>

// Problem constants (B, SLIDE, L, D) = (4, 16, 128, 64)
#define BDIM  4
#define SLD   16
#define LDIM  128
#define DDIM  64
#define PSTR  65   // padded LDS row stride (floats) -> conflict-free column walks

typedef _Float16 v16h __attribute__((ext_vector_type(16)));
typedef float    v8f  __attribute__((ext_vector_type(8)));

// ---- WMMA fragment builders (ISA 7.12.2, 16-bit layouts, wave32) -----------
// A 16x32 (MxK): lane half 0 -> K base 0, half 1 -> K base +16;
//   element e -> K = base+e (matches B fragment order).
__device__ __forceinline__ v16h load_A_frag(const float* __restrict__ ec,
                                            int rowBase, int kc, int lane) {
    const int m    = rowBase + (lane & 15);
    const int half = (lane >> 4) & 1;
    const float* src = ec + m * DDIM + 32 * kc + half * 16;
    v16h a;
#pragma unroll
    for (int e = 0; e < 16; ++e) {
        const int k = e;
        a[e] = (_Float16)src[k];
    }
    return a;
}

// B 32x16 (KxN): lanes 0-15 hold K 0..15, lanes 16-31 hold K 16..31;
//   element e is simply K = half*16 + e.  B[k][n] = W[n][k]  (for Ec @ W^T).
__device__ __forceinline__ v16h load_B_frag(const float* __restrict__ W,
                                            int ntile, int kc, int lane) {
    const int n    = ntile * 16 + (lane & 15);
    const int half = (lane >> 4) & 1;
    const float* src = W + n * DDIM + 32 * kc + half * 16;
    v16h b;
#pragma unroll
    for (int e = 0; e < 16; ++e) b[e] = (_Float16)src[e];
    return b;
}

__device__ __forceinline__ v8f wmma_f16(v16h a, v16h b, v8f c) {
    // (neg_a, A, neg_b, B, c_mod, C, reuse_a, reuse_b)
    return __builtin_amdgcn_wmma_f32_16x16x32_f16(false, a, false, b,
                                                  (short)0, c, false, false);
}

// tanh(x) = 1 - 2/(exp(2x)+1)   via v_exp_f32 + v_rcp_f32 (precision is
// provably irrelevant to the final output: softmax weights sum to 1).
__device__ __forceinline__ float fast_tanh(float x) {
    const float t = __builtin_amdgcn_exp2f(x * 2.8853900817779268f); // 2*log2(e)
    return 1.0f - 2.0f * __builtin_amdgcn_rcpf(t + 1.0f);
}

__global__ __launch_bounds__(256)
void attn_fused(const float* __restrict__ Ec, const float* __restrict__ Wi,
                const float* __restrict__ Wj, const float* __restrict__ V,
                float* __restrict__ out) {
    __shared__ float Plds[LDIM * PSTR];   // P[b,s]  = Ec[b,s]   @ Wi^T
    __shared__ float Qlds[LDIM * PSTR];   // Q[c,s]  = Ec[c,s]   @ Wj^T
    __shared__ float Vsh[DDIM];

    const int tid  = threadIdx.x;
    const int lane = tid & 31;
    const int wave = tid >> 5;
    const int blk  = blockIdx.x;
    const int c = blk & 3;            // = i & 3 for all rows this block owns
    const int s = (blk >> 2) & 15;
    const int b = blk >> 6;

    if (tid < DDIM) Vsh[tid] = V[tid];

    const float* EcP = Ec + (size_t)((b * SLD + s) * LDIM) * DDIM;
    const float* EcQ = Ec + (size_t)((c * SLD + s) * LDIM) * DDIM;

    __builtin_prefetch(EcP + wave * 16 * DDIM, 0, 1);
    __builtin_prefetch(EcQ + wave * 16 * DDIM, 0, 1);

    // -------- Stage 1: dual GEMM into LDS via WMMA (wave = 16-row strip) ----
    {
        const int rowBase = wave * 16;
        const v16h aP0 = load_A_frag(EcP, rowBase, 0, lane);
        const v16h aP1 = load_A_frag(EcP, rowBase, 1, lane);
        const v16h aQ0 = load_A_frag(EcQ, rowBase, 0, lane);
        const v16h aQ1 = load_A_frag(EcQ, rowBase, 1, lane);
        const int nloc  = lane & 15;
        const int rhalf = (lane >> 4) * 8;   // C/D: VGPR r -> M = r + 8*half
#pragma unroll
        for (int t = 0; t < 4; ++t) {
            v8f accP = {};
            accP = wmma_f16(aP0, load_B_frag(Wi, t, 0, lane), accP);
            accP = wmma_f16(aP1, load_B_frag(Wi, t, 1, lane), accP);
            v8f accQ = {};
            accQ = wmma_f16(aQ0, load_B_frag(Wj, t, 0, lane), accQ);
            accQ = wmma_f16(aQ1, load_B_frag(Wj, t, 1, lane), accQ);
            const int n = t * 16 + nloc;
#pragma unroll
            for (int r = 0; r < 8; ++r) {
                const int row = rowBase + rhalf + r;
                Plds[row * PSTR + n] = accP[r];
                Qlds[row * PSTR + n] = accQ[r];
            }
        }
    }
    __syncthreads();

    // -------- Stage 2: scores + softmax + (sum alpha) * Ec ------------------
    const float LOG2E = 1.4426950408889634f;
    const int jq = lane & 3;              // == j & 3 for all of this lane's j
#pragma unroll 1
    for (int u = 0; u < 4; ++u) {
        const int i = c + 4 * (wave * 4 + u);        // this block: i % 4 == c
        const float* Prow = &Plds[(((i & ~3) | jq)) * PSTR]; // P[b,s,(i&~3)|(j&3)]
        const float* Q0 = &Qlds[(lane +  0) * PSTR];         // Q[i&3,s,j]
        const float* Q1 = &Qlds[(lane + 32) * PSTR];
        const float* Q2 = &Qlds[(lane + 64) * PSTR];
        const float* Q3 = &Qlds[(lane + 96) * PSTR];
        float a0 = 0.f, a1 = 0.f, a2 = 0.f, a3 = 0.f;
#pragma unroll 4
        for (int dd = 0; dd < DDIM; ++dd) {
            const float vv = Vsh[dd];
            const float p  = Prow[dd];
            a0 = fmaf(vv, fast_tanh(p + Q0[dd]), a0);
            a1 = fmaf(vv, fast_tanh(p + Q1[dd]), a1);
            a2 = fmaf(vv, fast_tanh(p + Q2[dd]), a2);
            a3 = fmaf(vv, fast_tanh(p + Q3[dd]), a3);
        }
        // softmax over the 128 j's (4 per lane x 32 lanes), then sum(alpha)
        float mx = fmaxf(fmaxf(a0, a1), fmaxf(a2, a3));
#pragma unroll
        for (int off = 16; off >= 1; off >>= 1)
            mx = fmaxf(mx, __shfl_xor(mx, off, 32));
        const float e0 = __builtin_amdgcn_exp2f((a0 - mx) * LOG2E);
        const float e1 = __builtin_amdgcn_exp2f((a1 - mx) * LOG2E);
        const float e2 = __builtin_amdgcn_exp2f((a2 - mx) * LOG2E);
        const float e3 = __builtin_amdgcn_exp2f((a3 - mx) * LOG2E);
        const float zloc = (e0 + e1) + (e2 + e3);
        float Z = zloc;
#pragma unroll
        for (int off = 16; off >= 1; off >>= 1) Z += __shfl_xor(Z, off, 32);
        float SA = zloc * __builtin_amdgcn_rcpf(Z);   // sum of this lane's alphas
#pragma unroll
        for (int off = 16; off >= 1; off >>= 1) SA += __shfl_xor(SA, off, 32);
        // attn row = Ec row * sum_j(alpha)   (== Ec, up to fp rounding)
        const size_t m = (size_t)((b * SLD + s) * LDIM + i);
        const float* er = Ec  + m * DDIM;
        float*     orow = out + m * DDIM;
        orow[lane]      = er[lane]      * SA;
        orow[lane + 32] = er[lane + 32] * SA;
    }
}

extern "C" void kernel_launch(void* const* d_in, const int* in_sizes, int n_in,
                              void* d_out, int out_size, void* d_ws, size_t ws_size,
                              hipStream_t stream) {
    (void)in_sizes; (void)n_in; (void)d_ws; (void)ws_size; (void)out_size;
    const float* Ec = (const float*)d_in[0];
    const float* Wi = (const float*)d_in[1];
    const float* Wj = (const float*)d_in[2];
    const float* V  = (const float*)d_in[3];
    float* out = (float*)d_out;
    // 256 blocks = (b, s, c=i&3); 256 threads = 8 wave32s per block.
    attn_fused<<<dim3(BDIM * SLD * 4), dim3(256), 0, stream>>>(Ec, Wi, Wj, V, out);
}
